// SolvEncoder_19181323944367
// MI455X (gfx1250) — matrix operand order repaired and checked by reading, not compiled, not run on hardware
//
#include <hip/hip_runtime.h>
#include <hip/hip_bf16.h>
#include <stddef.h>
#include <stdint.h>

// ---------------- constants (match reference) ----------------
#define BB     4
#define NN     2048
#define KK     12
#define DIMH   128
#define MDIM   16
#define FEATD  10
#define DEPTH  4
#define CLAMPV 2.0f
#define EICOL  514          // 2*EI output cols of edge GEMM1 (EI=257)
#define NT1    33           // ceil(514/16) -> 528 cols padded
#define KT1    8            // 256 / 32
#define KT2    17           // 544 / 32 (514 padded)
#define NKT    5            // node GEMM1 K: 144 -> 160 / 32
#define NNT1   16           // node GEMM1 N: 256/16
#define NKT2   8            // node GEMM2 K: 256/32
#define NNT2   8            // node GEMM2 N: 128/16

typedef __attribute__((ext_vector_type(16))) _Float16 v16h;
typedef __attribute__((ext_vector_type(8)))  float    v8f;

union HFrag { v16h v; uint4 q[2]; };

static __device__ __forceinline__ v8f vzero8() {
  v8f z;
#pragma unroll
  for (int i = 0; i < 8; ++i) z[i] = 0.f;
  return z;
}
static __device__ __forceinline__ float silu_f(float x) {
  return x / (1.f + __expf(-x));
}

// Async global->LDS 16-byte copy (CDNA5, tracked by ASYNCcnt).
// LDS byte address comes from the lds_off VGPR; global = SGPR base + VGPR byte offset (GVS mode).
static __device__ __forceinline__ void async_copy_b128(unsigned lds_off,
                                                       const _Float16* base,
                                                       unsigned byte_off) {
  asm volatile("global_load_async_to_lds_b128 %0, %1, %2"
               :
               : "v"(lds_off), "v"(byte_off), "s"(base)
               : "memory");
}
static __device__ __forceinline__ void wait_async0() {
  asm volatile("s_wait_asynccnt 0x0" ::: "memory");
}
static __device__ __forceinline__ unsigned lds_off_of(const void* p) {
  // generic-pointer low 32 bits == LDS byte offset (aperture mapping)
  return (unsigned)(uintptr_t)p;
}

// ---------------- weight fragment packer ----------------
// p = ((kt*NT + nt)*32 + lane)*16 + t
// lane<16 holds K rows kt*32 + t ; lane>=16 holds kt*32 + 16 + t ; col = nt*16 + (lane&15)
__global__ void pack_w_kernel(const float* __restrict__ src, _Float16* __restrict__ dst,
                              int Ksrc, int Nsrc, int KT, int NT,
                              int srcLayerStride, int dstLayerStride) {
  int l = blockIdx.y;
  int p = blockIdx.x * 256 + threadIdx.x;
  int total = KT * NT * 512;
  if (p >= total) return;
  int t    = p & 15;
  int lane = (p >> 4) & 31;
  int nt   = (p >> 9) % NT;
  int kt   = p / (NT * 512);
  int col  = nt * 16 + (lane & 15);
  int krow = kt * 32 + ((lane < 16) ? t : 16 + t);
  float v = (krow < Ksrc && col < Nsrc) ? src[(size_t)l * srcLayerStride + (size_t)krow * Nsrc + col] : 0.f;
  dst[(size_t)l * dstLayerStride + p] = (_Float16)v;
}

// ---------------- feature embedding (writes f32 + f16 mirror) ----------------
__global__ void embed_kernel(const float* __restrict__ feats, const float* __restrict__ emb_w,
                             const float* __restrict__ emb_b, float* __restrict__ h0,
                             _Float16* __restrict__ h0f) {
  int idx = blockIdx.x * 256 + threadIdx.x;           // BB*NN*DIMH
  if (idx >= BB * NN * DIMH) return;
  int node = idx >> 7;
  int d = idx & 127;
  float s = emb_b[d];
#pragma unroll
  for (int f = 0; f < FEATD; ++f) s += feats[(size_t)node * FEATD + f] * emb_w[f * DIMH + d];
  h0[idx] = s;
  h0f[idx] = (_Float16)s;
}

// ---------------- kNN (top-12 smallest masked dist, stable by index) ----------------
__global__ __launch_bounds__(64) void knn_kernel(const float* __restrict__ coords,
                                                 const unsigned char* __restrict__ mask,
                                                 int* __restrict__ nbhd, float* __restrict__ distk) {
  __shared__ float SD[2][32][13];
  __shared__ int   SI[2][32][13];
  int wave = threadIdx.x >> 5, lane = threadIdx.x & 31;
  int node = blockIdx.x * 2 + wave;
  int b = node >> 11, i = node & (NN - 1);
  const float* cb = coords + (size_t)b * NN * 3;
  float cix = cb[i * 3 + 0], ciy = cb[i * 3 + 1], ciz = cb[i * 3 + 2];
  bool mi = mask[(size_t)b * NN + i] != 0;

  float bd[12]; int bj[12];
#pragma unroll
  for (int q = 0; q < 12; ++q) { bd[q] = 3.0e38f; bj[q] = 0x7fffffff; }

  for (int t = 0; t < NN / 32; ++t) {
    int j = lane + t * 32;
    float dx = cix - cb[j * 3 + 0], dy = ciy - cb[j * 3 + 1], dz = ciz - cb[j * 3 + 2];
    float d = dx * dx + dy * dy + dz * dz;
    bool pm = mi && (mask[(size_t)b * NN + j] != 0);
    float rank = pm ? d : 1e5f;
    bool ins = (rank < bd[11]) || (rank == bd[11] && j < bj[11]);
    if (ins) {
      float cr = rank; int cj = j;
#pragma unroll
      for (int q = 0; q < 12; ++q) {
        bool lt = (cr < bd[q]) || (cr == bd[q] && cj < bj[q]);
        if (lt) { float td = bd[q]; int ti = bj[q]; bd[q] = cr; bj[q] = cj; cr = td; cj = ti; }
      }
    }
  }
#pragma unroll
  for (int q = 0; q < 12; ++q) { SD[wave][lane][q] = bd[q]; SI[wave][lane][q] = bj[q]; }
  SD[wave][lane][12] = 3.0e38f; SI[wave][lane][12] = 0x7fffffff;

  for (int s = 1; s < 32; s <<= 1) {
    __syncthreads();
    if ((lane & (2 * s - 1)) == 0) {
      float od[12]; int oi[12];
      int pa = 0, pb = 0;
#pragma unroll
      for (int q = 0; q < 12; ++q) {
        float da = SD[wave][lane][pa];     int ia = SI[wave][lane][pa];
        float db = SD[wave][lane + s][pb]; int ib = SI[wave][lane + s][pb];
        bool ta = (da < db) || (da == db && ia < ib);
        od[q] = ta ? da : db; oi[q] = ta ? ia : ib;
        if (ta) pa++; else pb++;
      }
#pragma unroll
      for (int q = 0; q < 12; ++q) { SD[wave][lane][q] = od[q]; SI[wave][lane][q] = oi[q]; }
    }
  }
  __syncthreads();
  if (lane < KK) {
    int j = SI[wave][0][lane];
    nbhd[(size_t)node * KK + lane] = j;
    float dx = cix - cb[j * 3 + 0], dy = ciy - cb[j * 3 + 1], dz = ciz - cb[j * 3 + 2];
    distk[(size_t)node * KK + lane] = dx * dx + dy * dy + dz * dz;   // true sq-dist
  }
}

// ---------------- edge MLP + coor MLP (WMMA core, async-staged A) ----------------
// block = 64 threads = 2 waves; each wave handles 16 edges.
__global__ __launch_bounds__(64) void edge_mlp_kernel(
    const _Float16* __restrict__ hf, const int* __restrict__ nbhd, const float* __restrict__ distk,
    const unsigned char* __restrict__ mask,
    const _Float16* __restrict__ ew1b, const float* __restrict__ ew1L, const float* __restrict__ eb1L,
    const _Float16* __restrict__ ew2b, const float* __restrict__ eb2L,
    const float* __restrict__ cw1L, const float* __restrict__ cb1L,
    const float* __restrict__ cw2L, const float* __restrict__ cb2L,
    float* __restrict__ mij, float* __restrict__ wco) {
  __shared__ __align__(16) _Float16 Abuf[2][16][264];   // 16 edges x 256 (+8 pad)
  __shared__ __align__(16) _Float16 Hbuf[2][16][552];   // 16 edges x 544 (+8 pad)
  __shared__ float Dist[2][16];
  __shared__ int   Jsh[2][16];
  __shared__ float Mbuf[2][16][17];
  __shared__ float Wpart[2][32];

  int wave = threadIdx.x >> 5, lane = threadIdx.x & 31;
  int e0 = (blockIdx.x * 2 + wave) * 16;

  if (lane < 16) {
    int e = e0 + lane;
    Jsh[wave][lane]  = nbhd[e];
    Dist[wave][lane] = distk[e];
  }
  __syncthreads();

  // ---- async stage A: lane<16 copies h_i chunk, lane>=16 copies h_j chunk (16B each)
  {
    unsigned abase = lds_off_of(&Abuf[wave][0][0]);
    int chunk = lane & 15;
#pragma unroll
    for (int ed = 0; ed < 16; ++ed) {
      int e = e0 + ed;
      int b = e / (NN * KK);
      int i = (e / KK) & (NN - 1);
      int row = (lane < 16) ? (b * NN + i) : (b * NN + Jsh[wave][ed]);
      unsigned goff = (unsigned)row * (DIMH * 2) + (unsigned)chunk * 16;
      unsigned loff = abase + (unsigned)ed * (264 * 2) + ((lane < 16) ? 0u : 256u) + (unsigned)chunk * 16;
      async_copy_b128(loff, hf, goff);
    }
  }
  // zero Hbuf pad columns [528,552) while async copies fly
  for (int idx = lane; idx < 16 * 24; idx += 32) {
    int r = idx / 24, c = 528 + idx % 24;
    Hbuf[wave][r][c] = (_Float16)0.f;
  }
  wait_async0();
  __syncthreads();

  int rowL = lane & 15;
  int mb = (lane < 16) ? 0 : 8;
  int aoff = (lane < 16) ? 0 : 16;

  // ---- GEMM1: (16 x 256) @ (256 x 528) + rank-1 dist term, silu -> Hbuf
#pragma unroll 1
  for (int nt = 0; nt < NT1; ++nt) {
    v8f acc = vzero8();
#pragma unroll
    for (int kt = 0; kt < KT1; ++kt) {
      HFrag Af, Bf;
      const uint4* ap = (const uint4*)&Abuf[wave][rowL][kt * 32 + aoff];
      Af.q[0] = ap[0]; Af.q[1] = ap[1];
      const uint4* bp = (const uint4*)(ew1b + (((size_t)(kt * NT1 + nt) * 32 + lane) * 16));
      Bf.q[0] = bp[0]; Bf.q[1] = bp[1];
      acc = __builtin_amdgcn_wmma_f32_16x16x32_f16(false, Af.v, false, Bf.v, (short)0, acc, false, false);
    }
    int col = nt * 16 + rowL;
    float w256 = (col < EICOL) ? ew1L[256 * EICOL + col] : 0.f;
    float bias = (col < EICOL) ? eb1L[col] : 0.f;
#pragma unroll
    for (int r = 0; r < 8; ++r) {
      float x = acc[r] + bias + Dist[wave][mb + r] * w256;
      Hbuf[wave][mb + r][col] = (_Float16)silu_f(x);
    }
  }
  __syncthreads();

  // ---- GEMM2: (16 x 544) @ (544 x 16), silu -> m (16x16)
  {
    v8f acc = vzero8();
#pragma unroll
    for (int kt = 0; kt < KT2; ++kt) {
      HFrag Af, Bf;
      const uint4* ap = (const uint4*)&Hbuf[wave][rowL][kt * 32 + aoff];
      Af.q[0] = ap[0]; Af.q[1] = ap[1];
      const uint4* bp = (const uint4*)(ew2b + (((size_t)kt * 32 + lane) * 16));
      Bf.q[0] = bp[0]; Bf.q[1] = bp[1];
      acc = __builtin_amdgcn_wmma_f32_16x16x32_f16(false, Af.v, false, Bf.v, (short)0, acc, false, false);
    }
    float bias = eb2L[rowL];
#pragma unroll
    for (int r = 0; r < 8; ++r) {
      Mbuf[wave][mb + r][rowL] = silu_f(acc[r] + bias);
    }
  }
  __syncthreads();

  // store m_ij
  for (int idx = lane; idx < 256; idx += 32) {
    int ed = idx >> 4, c = idx & 15;
    mij[(size_t)(e0 + ed) * MDIM + c] = Mbuf[wave][ed][c];
  }

  // ---- coor MLP: w = silu(m @ cw1 + cb1) @ cw2 + cb2 (split over 2 half-lanes)
  {
    int half = lane >> 4;
    int m = lane & 15;
    float acc = 0.f;
    for (int j0 = 0; j0 < 32; ++j0) {
      int j = half * 32 + j0;
      float hs = cb1L[j];
#pragma unroll
      for (int c = 0; c < MDIM; ++c) hs += Mbuf[wave][m][c] * cw1L[c * 64 + j];
      acc += silu_f(hs) * cw2L[j];
    }
    Wpart[wave][lane] = acc;
  }
  __syncthreads();
  if (lane < 16) {
    int e = e0 + lane;
    int b = e / (NN * KK);
    int i = (e / KK) & (NN - 1);
    int j = Jsh[wave][lane];
    bool m2 = (mask[(size_t)b * NN + i] != 0) && (mask[(size_t)b * NN + j] != 0);
    float wv = Wpart[wave][lane] + Wpart[wave][lane + 16] + cb2L[0];
    wv = m2 ? wv : 0.f;
    wv = fminf(fmaxf(wv, -CLAMPV), CLAMPV);
    wco[e] = wv;
  }
}

// ---------------- coordinate update + message aggregation ----------------
__global__ void coord_agg_kernel(const float* __restrict__ cin, const int* __restrict__ nbhd,
                                 const float* __restrict__ wco, const float* __restrict__ mij,
                                 const unsigned char* __restrict__ mask, const float* __restrict__ cscaleL,
                                 float* __restrict__ cout, _Float16* __restrict__ mi_out) {
  int node = blockIdx.x * 256 + threadIdx.x;
  if (node >= BB * NN) return;
  int b = node >> 11, i = node & (NN - 1);
  const float* cb = cin + (size_t)b * NN * 3;
  float cx = cb[i * 3 + 0], cy = cb[i * 3 + 1], cz = cb[i * 3 + 2];
  float cs = cscaleL[0];
  bool mi_ = mask[(size_t)b * NN + i] != 0;
  float ax = 0.f, ay = 0.f, az = 0.f;
  float macc[MDIM];
#pragma unroll
  for (int c = 0; c < MDIM; ++c) macc[c] = 0.f;
  for (int k = 0; k < KK; ++k) {
    size_t e = (size_t)node * KK + k;
    int j = nbhd[e];
    float rx = cx - cb[j * 3 + 0], ry = cy - cb[j * 3 + 1], rz = cz - cb[j * 3 + 2];
    float nrm = sqrtf(rx * rx + ry * ry + rz * rz);
    float inv = cs / fmaxf(nrm, 1e-8f);
    float wv = wco[e];
    ax += wv * rx * inv; ay += wv * ry * inv; az += wv * rz * inv;
    bool m2 = mi_ && (mask[(size_t)b * NN + j] != 0);
    if (m2) {
#pragma unroll
      for (int c = 0; c < MDIM; ++c) macc[c] += mij[e * MDIM + c];
    }
  }
  float* co = cout + (size_t)b * NN * 3;
  co[i * 3 + 0] = cx + ax; co[i * 3 + 1] = cy + ay; co[i * 3 + 2] = cz + az;
#pragma unroll
  for (int c = 0; c < MDIM; ++c) mi_out[(size_t)node * MDIM + c] = (_Float16)macc[c];
}

// ---------------- node MLP (WMMA, async-staged A) ----------------
__global__ __launch_bounds__(64) void node_mlp_kernel(
    const float* __restrict__ hin, const _Float16* __restrict__ hinf,
    const _Float16* __restrict__ mif,
    const _Float16* __restrict__ nw1b, const float* __restrict__ nb1L,
    const _Float16* __restrict__ nw2b, const float* __restrict__ nb2L,
    float* __restrict__ hout, _Float16* __restrict__ houtf) {
  __shared__ __align__(16) _Float16 A2[2][16][168];    // 16 nodes x 160 (+8 pad)
  __shared__ __align__(16) _Float16 H2[2][16][264];    // 16 nodes x 256 (+8 pad)
  int wave = threadIdx.x >> 5, lane = threadIdx.x & 31;
  int n0 = (blockIdx.x * 2 + wave) * 16;
  unsigned a2base = lds_off_of(&A2[wave][0][0]);

  // async stage: h rows (16 nodes x 256B) -> 8 instrs; mi rows (16 x 32B) -> 1 instr
#pragma unroll
  for (int it = 0; it < 8; ++it) {
    int idx = it * 32 + lane;
    int nd = idx >> 4, ch = idx & 15;
    unsigned goff = (unsigned)(n0 + nd) * (DIMH * 2) + (unsigned)ch * 16;
    unsigned loff = a2base + (unsigned)nd * (168 * 2) + (unsigned)ch * 16;
    async_copy_b128(loff, hinf, goff);
  }
  {
    int nd = lane >> 1, ch = lane & 1;
    unsigned goff = (unsigned)(n0 + nd) * (MDIM * 2) + (unsigned)ch * 16;
    unsigned loff = a2base + (unsigned)nd * (168 * 2) + 256u + (unsigned)ch * 16;
    async_copy_b128(loff, mif, goff);
  }
  // zero pad cols [144,160) while copies fly
  for (int idx = lane; idx < 16 * 16; idx += 32) {
    int nd = idx >> 4, c = 144 + (idx & 15);
    A2[wave][nd][c] = (_Float16)0.f;
  }
  // zero H2 pad columns [256,264)
  for (int idx = lane; idx < 16 * 8; idx += 32) {
    int r = idx / 8, c = 256 + idx % 8;
    H2[wave][r][c] = (_Float16)0.f;
  }
  wait_async0();
  __syncthreads();

  int rowL = lane & 15;
  int mb = (lane < 16) ? 0 : 8;
  int aoff = (lane < 16) ? 0 : 16;

#pragma unroll 1
  for (int nt = 0; nt < NNT1; ++nt) {
    v8f acc = vzero8();
#pragma unroll
    for (int kt = 0; kt < NKT; ++kt) {
      HFrag Af, Bf;
      const uint4* ap = (const uint4*)&A2[wave][rowL][kt * 32 + aoff];
      Af.q[0] = ap[0]; Af.q[1] = ap[1];
      const uint4* bp = (const uint4*)(nw1b + (((size_t)(kt * NNT1 + nt) * 32 + lane) * 16));
      Bf.q[0] = bp[0]; Bf.q[1] = bp[1];
      acc = __builtin_amdgcn_wmma_f32_16x16x32_f16(false, Af.v, false, Bf.v, (short)0, acc, false, false);
    }
    int col = nt * 16 + rowL;
    float bias = nb1L[col];
#pragma unroll
    for (int r = 0; r < 8; ++r) {
      H2[wave][mb + r][col] = (_Float16)silu_f(acc[r] + bias);
    }
  }
  __syncthreads();

#pragma unroll 1
  for (int nt = 0; nt < NNT2; ++nt) {
    v8f acc = vzero8();
#pragma unroll
    for (int kt = 0; kt < NKT2; ++kt) {
      HFrag Af, Bf;
      const uint4* ap = (const uint4*)&H2[wave][rowL][kt * 32 + aoff];
      Af.q[0] = ap[0]; Af.q[1] = ap[1];
      const uint4* bp = (const uint4*)(nw2b + (((size_t)(kt * NNT2 + nt) * 32 + lane) * 16));
      Bf.q[0] = bp[0]; Bf.q[1] = bp[1];
      acc = __builtin_amdgcn_wmma_f32_16x16x32_f16(false, Af.v, false, Bf.v, (short)0, acc, false, false);
    }
    int col = nt * 16 + rowL;
    float bias = nb2L[col];
#pragma unroll
    for (int r = 0; r < 8; ++r) {
      int node = n0 + mb + r;
      float res = hin[(size_t)node * DIMH + col];
      float v = acc[r] + bias + res;
      hout[(size_t)node * DIMH + col] = v;
      houtf[(size_t)node * DIMH + col] = (_Float16)v;
    }
  }
}

// ---------------- masked mean pool ----------------
__global__ void pool_kernel(const float* __restrict__ h, const unsigned char* __restrict__ mask,
                            float* __restrict__ out) {
  int b = blockIdx.x;
  int d = threadIdx.x;   // 128
  float s = 0.f, cnt = 0.f;
  for (int n = 0; n < NN; ++n) {
    float mf = (mask[(size_t)b * NN + n] != 0) ? 1.f : 0.f;
    s += h[((size_t)b * NN + n) * DIMH + d] * mf;
    cnt += mf;
  }
  out[b * DIMH + d] = s / cnt;
}

// ---------------- host-side orchestration ----------------
extern "C" void kernel_launch(void* const* d_in, const int* in_sizes, int n_in,
                              void* d_out, int out_size, void* d_ws, size_t ws_size,
                              hipStream_t stream) {
  (void)in_sizes; (void)n_in; (void)out_size; (void)ws_size;
  const float* feats  = (const float*)d_in[0];
  const float* coords = (const float*)d_in[1];
  const unsigned char* mask = (const unsigned char*)d_in[2];
  const float* emb_w = (const float*)d_in[3];
  const float* emb_b = (const float*)d_in[4];
  const float* ew1 = (const float*)d_in[5];
  const float* eb1 = (const float*)d_in[6];
  const float* ew2 = (const float*)d_in[7];
  const float* eb2 = (const float*)d_in[8];
  const float* cw1 = (const float*)d_in[9];
  const float* cb1 = (const float*)d_in[10];
  const float* cw2 = (const float*)d_in[11];
  const float* cb2 = (const float*)d_in[12];
  const float* cscale = (const float*)d_in[13];
  const float* nw1 = (const float*)d_in[14];
  const float* nb1 = (const float*)d_in[15];
  const float* nw2 = (const float*)d_in[16];
  const float* nb2 = (const float*)d_in[17];

  char* ws = (char*)d_ws;
  size_t off = 0;
  float* h0   = (float*)(ws + off); off += (size_t)BB * NN * DIMH * 4;       // 4 MB
  float* h1   = (float*)(ws + off); off += (size_t)BB * NN * DIMH * 4;       // 4 MB
  _Float16* hf0 = (_Float16*)(ws + off); off += (size_t)BB * NN * DIMH * 2;  // 2 MB
  _Float16* hf1 = (_Float16*)(ws + off); off += (size_t)BB * NN * DIMH * 2;  // 2 MB
  float* c0   = (float*)(ws + off); off += (size_t)BB * NN * 3 * 4;
  float* c1   = (float*)(ws + off); off += (size_t)BB * NN * 3 * 4;
  int*   nbh  = (int*)  (ws + off); off += (size_t)BB * NN * KK * 4;
  float* dstk = (float*)(ws + off); off += (size_t)BB * NN * KK * 4;
  float* mijv = (float*)(ws + off); off += (size_t)BB * NN * KK * MDIM * 4;  // 6 MB
  float* wcov = (float*)(ws + off); off += (size_t)BB * NN * KK * 4;
  _Float16* mif = (_Float16*)(ws + off); off += (size_t)BB * NN * MDIM * 2;
  const int EW1B_L = KT1 * NT1 * 512;   // 135168
  const int EW2B_L = KT2 * 1 * 512;     // 8704
  const int NW1B_L = NKT * NNT1 * 512;  // 40960
  const int NW2B_L = NKT2 * NNT2 * 512; // 32768
  _Float16* ew1b = (_Float16*)(ws + off); off += (size_t)DEPTH * EW1B_L * 2;
  _Float16* ew2b = (_Float16*)(ws + off); off += (size_t)DEPTH * EW2B_L * 2;
  _Float16* nw1b = (_Float16*)(ws + off); off += (size_t)DEPTH * NW1B_L * 2;
  _Float16* nw2b = (_Float16*)(ws + off); off += (size_t)DEPTH * NW2B_L * 2;

  // pack weights (every call; deterministic, cheap)
  pack_w_kernel<<<dim3((EW1B_L + 255) / 256, DEPTH), 256, 0, stream>>>(ew1, ew1b, 257, EICOL, KT1, NT1, 257 * EICOL, EW1B_L);
  pack_w_kernel<<<dim3((EW2B_L + 255) / 256, DEPTH), 256, 0, stream>>>(ew2, ew2b, EICOL, MDIM, KT2, 1, EICOL * MDIM, EW2B_L);
  pack_w_kernel<<<dim3((NW1B_L + 255) / 256, DEPTH), 256, 0, stream>>>(nw1, nw1b, DIMH + MDIM, 2 * DIMH, NKT, NNT1, (DIMH + MDIM) * 2 * DIMH, NW1B_L);
  pack_w_kernel<<<dim3((NW2B_L + 255) / 256, DEPTH), 256, 0, stream>>>(nw2, nw2b, 2 * DIMH, DIMH, NKT2, NNT2, 2 * DIMH * DIMH, NW2B_L);

  embed_kernel<<<(BB * NN * DIMH) / 256, 256, 0, stream>>>(feats, emb_w, emb_b, h0, hf0);

  const float* hc = h0; float* hn = h1;
  const _Float16* hfc = hf0; _Float16* hfn = hf1;
  const float* cc = coords;
  float* cbufs[2] = {c0, c1};
  for (int l = 0; l < DEPTH; ++l) {
    float* cn = cbufs[l & 1];
    knn_kernel<<<(BB * NN) / 2, 64, 0, stream>>>(cc, mask, nbh, dstk);
    edge_mlp_kernel<<<(BB * NN * KK) / 32, 64, 0, stream>>>(
        hfc, nbh, dstk, mask,
        ew1b + (size_t)l * EW1B_L, ew1 + (size_t)l * 257 * EICOL, eb1 + (size_t)l * EICOL,
        ew2b + (size_t)l * EW2B_L, eb2 + (size_t)l * MDIM,
        cw1 + (size_t)l * MDIM * 64, cb1 + (size_t)l * 64,
        cw2 + (size_t)l * 64, cb2 + l,
        mijv, wcov);
    coord_agg_kernel<<<(BB * NN) / 256, 256, 0, stream>>>(cc, nbh, wcov, mijv, mask, cscale + l, cn, mif);
    node_mlp_kernel<<<(BB * NN) / 32, 64, 0, stream>>>(
        hc, hfc, mif,
        nw1b + (size_t)l * NW1B_L, nb1 + (size_t)l * 2 * DIMH,
        nw2b + (size_t)l * NW2B_L, nb2 + (size_t)l * DIMH,
        hn, hfn);
    // ping-pong
    const float* th = hc; hc = hn; hn = (float*)th;
    const _Float16* tf = hfc; hfc = hfn; hfn = (_Float16*)tf;
    cc = cn;
  }
  pool_kernel<<<BB, DIMH, 0, stream>>>(hc, mask, (float*)d_out);
}
